// CausalSelfAttention_3633542333180
// MI455X (gfx1250) — matrix operand order repaired and checked by reading, not compiled, not run on hardware
//
#include <hip/hip_runtime.h>
#include <hip/hip_bf16.h>

// ---------------------------------------------------------------------------
// CDNA5 (gfx1250, wave32) causal self-attention forward.
//  Pre-pass: x -> bf16, W_qkv/W_proj -> bf16 transposed [N][K] so GEMM inner
//  loops have zero conversion VALU work.  All tile staging uses
//  global_load_async_to_lds_b128 with a 2x-unrolled ping-pong pipeline:
//  buffer indices are compile-time constants, waits are s_wait_asynccnt 5/0 (GEMM) or 4/0 (attention)
//  (async loads complete in order), so tile k's DMA is done while tile k+1
//  streams behind the WMMAs.
//  qkv GEMM -> Q[B,H,T,hd], K[B,H,T,hd], Vt[B,H,hd,T] (bf16)
//  flash attention with v_wmma_f32_16x16x32_bf16
//  proj GEMM -> f32 out
// ---------------------------------------------------------------------------

typedef __attribute__((ext_vector_type(16))) __bf16 v16bf;
typedef __attribute__((ext_vector_type(8)))  float  v8f;

struct bf16x16 { uint4 lo, hi; };   // 32B == sizeof(v16bf)

#define D_MODEL   2048
#define N_HEADS   16
#define HEAD_DIM  128
#define T_SEQ     2048
#define BT_ROWS   4096              // B * T

__device__ __forceinline__ __bf16 f2bf(float f) { return (__bf16)f; }

__device__ __forceinline__ v8f wmma_bf16(v16bf a, v16bf b, v8f c) {
  return __builtin_amdgcn_wmma_f32_16x16x32_bf16(false, a, false, b,
                                                 (short)0, c, false, false);
}

// LDS byte address = low 32 bits of the generic pointer (ISA 10.2 aperture).
__device__ __forceinline__ unsigned lds_addr(const void* p) {
  return (unsigned)(unsigned long long)p;
}
// Async DMA: global -> LDS, 16 bytes per lane, tracked by ASYNCcnt.
__device__ __forceinline__ void async_ld_b128(unsigned lds, const void* gptr) {
  asm volatile("global_load_async_to_lds_b128 %0, %1, off"
               :: "v"(lds), "v"(gptr) : "memory");
}
template <int N>
__device__ __forceinline__ void wait_async() {
  asm volatile("s_wait_asynccnt %0" :: "n"(N) : "memory");
}

// ===========================================================================
// Pre-pass A: x (f32, row-major) -> bf16 row-major copy.
// ===========================================================================
__global__ __launch_bounds__(256)
void cvt_bf16_kernel(const float* __restrict__ X, __bf16* __restrict__ Xb)
{
  const size_t i = ((size_t)blockIdx.x * 256 + threadIdx.x) * 8;
  float4 a = *(const float4*)(X + i);
  float4 b = *(const float4*)(X + i + 4);
  __bf16 t[8] = { f2bf(a.x), f2bf(a.y), f2bf(a.z), f2bf(a.w),
                  f2bf(b.x), f2bf(b.y), f2bf(b.z), f2bf(b.w) };
  *(uint4*)(Xb + i) = *(const uint4*)t;
}

// ===========================================================================
// Pre-pass B: W (f32, [K][N]) -> Wt (bf16, [N][K]) via 32x32 LDS tiles.
// ===========================================================================
__global__ __launch_bounds__(256)
void cvt_transpose_kernel(const float* __restrict__ W, __bf16* __restrict__ Wt,
                          int K, int N)
{
  __shared__ __bf16 tile[32 * 36];              // [n][k], stride 36 (72B rows)
  const int tid = threadIdx.x;
  const int k0 = blockIdx.y * 32, n0 = blockIdx.x * 32;
  const int r = tid >> 3, c = tid & 7;
  float4 v = *(const float4*)(W + (size_t)(k0 + r) * N + n0 + c * 4);
  tile[(c * 4 + 0) * 36 + r] = f2bf(v.x);
  tile[(c * 4 + 1) * 36 + r] = f2bf(v.y);
  tile[(c * 4 + 2) * 36 + r] = f2bf(v.z);
  tile[(c * 4 + 3) * 36 + r] = f2bf(v.w);
  __syncthreads();
  *(uint2*)(Wt + (size_t)(n0 + r) * K + k0 + c * 4) =
      *(const uint2*)&tile[r * 36 + c * 4];
}

// ===========================================================================
// GEMM core: 256 thr = 8 waves (2M x 4N); wave tile 32x64 (8 WMMA);
// block tile 64(M) x 256(N), BK=32.  A,B both bf16, B pre-transposed [N][K].
// 2x-unrolled ping-pong pipeline, 5 async b128 per thread per tile.
// ===========================================================================
#define GEMM_BODY(...)                                                         \
  __shared__ __bf16 As[2][64 * 40];                                            \
  __shared__ __bf16 Bs[2][256 * 40];                                           \
  const int tid  = threadIdx.x;                                                \
  const int lane = tid & 31;                                                   \
  const int wave = tid >> 5;                                                   \
  const int lm   = lane & 15;                                                  \
  const int half = lane >> 4;                                                  \
  const int wm   = wave & 1;                                                   \
  const int wn   = wave >> 1;                                                  \
  const int rowBase = blockIdx.y * 64;                                         \
  const int colBase = blockIdx.x * 256;                                        \
  const v8f zero8 = {0.f,0.f,0.f,0.f,0.f,0.f,0.f,0.f};                         \
  v8f acc[2][4];                                                               \
  _Pragma("unroll") for (int i = 0; i < 2; i++)                                \
    _Pragma("unroll") for (int j = 0; j < 4; j++) acc[i][j] = zero8;           \
  const int sr = tid >> 2;                                                     \
  const int sc = tid & 3;                                                      \
  const __bf16* gA = A  + (size_t)(rowBase + sr) * D_MODEL + sc * 8;           \
  const __bf16* gB = Bt + (size_t)(colBase + sr) * D_MODEL + sc * 8;           \
  /* bb is always a literal -> constant LDS addresses, no movrels */           \
  auto issueT = [&](int k0, int bb) {                                          \
    async_ld_b128(lds_addr(&As[bb][sr * 40 + sc * 8]), gA + k0);               \
    _Pragma("unroll") for (int j = 0; j < 4; j++)                              \
      async_ld_b128(lds_addr(&Bs[bb][(sr + 64 * j) * 40 + sc * 8]),            \
                    gB + (size_t)(64 * j) * D_MODEL + k0);                     \
  };                                                                           \
  auto stepT = [&](int bb) {                                                   \
    v16bf afrag[2];                                                            \
    v16bf bfrag[4];                                                            \
    _Pragma("unroll") for (int mi = 0; mi < 2; mi++) {                         \
      const __bf16* ap = &As[bb][(wm * 32 + mi * 16 + lm) * 40 + half * 8];    \
      bf16x16 t; t.lo = *(const uint4*)ap; t.hi = *(const uint4*)(ap + 16);    \
      afrag[mi] = __builtin_bit_cast(v16bf, t);                                \
    }                                                                          \
    _Pragma("unroll") for (int ni = 0; ni < 4; ni++) {                         \
      const __bf16* bp = &Bs[bb][(wn * 64 + ni * 16 + lm) * 40 + half * 8];    \
      bf16x16 t; t.lo = *(const uint4*)bp; t.hi = *(const uint4*)(bp + 16);    \
      bfrag[ni] = __builtin_bit_cast(v16bf, t);                                \
    }                                                                          \
    _Pragma("unroll") for (int mi = 0; mi < 2; mi++)                           \
      _Pragma("unroll") for (int ni = 0; ni < 4; ni++)                         \
        acc[mi][ni] = wmma_bf16(afrag[mi], bfrag[ni], acc[mi][ni]);            \
  };                                                                           \
  issueT(0, 0);                                                                \
  issueT(32, 1);                                                               \
  for (int k0 = 0; k0 < D_MODEL - 64; k0 += 64) {                              \
    wait_async<5>(); __syncthreads(); stepT(0); __syncthreads();               \
    issueT(k0 + 64, 0);                                                        \
    wait_async<5>(); __syncthreads(); stepT(1); __syncthreads();               \
    issueT(k0 + 96, 1);                                                        \
  }                                                                            \
  wait_async<5>(); __syncthreads(); stepT(0); __syncthreads();                 \
  wait_async<0>(); __syncthreads(); stepT(1);                                  \
  __VA_ARGS__

// ===========================================================================
// Kernel 1: QKV GEMM.  Xb[4096,2048]bf16 @ Wt[6144,2048]bf16 + bias ->
// scatter Q/K (bf16 [B,H,T,hd]) and Vt (bf16 [B,H,hd,T]).
// ===========================================================================
__global__ __launch_bounds__(256)
void qkv_gemm_kernel(const __bf16* __restrict__ A, const __bf16* __restrict__ Bt,
                     const float* __restrict__ bias,
                     __bf16* __restrict__ Qo, __bf16* __restrict__ Ko,
                     __bf16* __restrict__ Vto)
{
  GEMM_BODY(
  {
    _Pragma("unroll") for (int mi = 0; mi < 2; mi++) {
      _Pragma("unroll") for (int ni = 0; ni < 4; ni++) {
        const int mbase = rowBase + wm * 32 + mi * 16 + 8 * half;
        const int col   = colBase + wn * 64 + ni * 16 + lm;
        const float bv  = bias[col];
        const int which = col >> 11;            // 0=Q 1=K 2=V
        const int cc    = col & 2047;
        const int h     = cc >> 7;
        const int d     = cc & 127;
        _Pragma("unroll") for (int r = 0; r < 8; r++) {
          const int m = mbase + r;              // global row in [0,4096)
          const int b = m >> 11;
          const int t = m & 2047;
          const size_t bh = (size_t)(b * N_HEADS + h);
          const __bf16 v = f2bf(acc[mi][ni][r] + bv);
          if (which == 0)      Qo [(bh * T_SEQ + t) * HEAD_DIM + d] = v;
          else if (which == 1) Ko [(bh * T_SEQ + t) * HEAD_DIM + d] = v;
          else                 Vto[(bh * HEAD_DIM + d) * T_SEQ + t] = v;
        }
      }
    }
  })
}

// ===========================================================================
// Kernel 3: proj GEMM.  AO[4096,2048]bf16 @ Wpt[2048,2048]bf16 + bias -> f32.
// ===========================================================================
__global__ __launch_bounds__(256)
void proj_gemm_kernel(const __bf16* __restrict__ A, const __bf16* __restrict__ Bt,
                      const float* __restrict__ bias, float* __restrict__ Out)
{
  GEMM_BODY(
  {
    _Pragma("unroll") for (int mi = 0; mi < 2; mi++) {
      _Pragma("unroll") for (int ni = 0; ni < 4; ni++) {
        const int mbase = rowBase + wm * 32 + mi * 16 + 8 * half;
        const int col   = colBase + wn * 64 + ni * 16 + lm;
        const float bv  = bias[col];
        _Pragma("unroll") for (int r = 0; r < 8; r++)
          Out[(size_t)(mbase + r) * D_MODEL + col] = acc[mi][ni][r] + bv;
      }
    }
  })
}

// ===========================================================================
// Kernel 2: flash attention.  Grid (qb=16, h=16, b=2), 256 thr = 8 waves,
// wave owns 16 q rows; 32-key blocks, 2x-unrolled ping-pong async staging;
// online softmax; O in 8 f32 fragments.
// ===========================================================================
__global__ __launch_bounds__(256)
void attn_kernel(const __bf16* __restrict__ Q, const __bf16* __restrict__ K,
                 const __bf16* __restrict__ Vt, __bf16* __restrict__ AO)
{
  __shared__ __bf16 Ks [2][32 * 136];    // [key][d]   stride 136 (272B rows)
  __shared__ __bf16 Vts[2][128 * 40];    // [d][key]   stride 40
  __shared__ __bf16 Ps [8][16 * 40];     // per-wave P [m][key]

  const int tid  = threadIdx.x;
  const int lane = tid & 31;
  const int wave = tid >> 5;
  const int lm   = lane & 15;
  const int half = lane >> 4;
  const int qb = blockIdx.x, h = blockIdx.y, b = blockIdx.z;
  const size_t bh = (size_t)(b * N_HEADS + h);
  const __bf16* Qh  = Q  + bh * T_SEQ * HEAD_DIM;
  const __bf16* Kh  = K  + bh * T_SEQ * HEAD_DIM;
  const __bf16* Vth = Vt + bh * HEAD_DIM * T_SEQ;

  const int qrow0 = qb * 128 + wave * 16;

  // Q fragments (A-layout, 4 chunks of K=32), loaded once from global
  v16bf qf[4];
  #pragma unroll
  for (int kc = 0; kc < 4; kc++) {
    const __bf16* qp = Qh + (size_t)(qrow0 + lm) * HEAD_DIM + kc * 32 + half * 8;
    bf16x16 t; t.lo = *(const uint4*)qp; t.hi = *(const uint4*)(qp + 16);
    qf[kc] = __builtin_bit_cast(v16bf, t);
  }

  const v8f zero8 = {0.f,0.f,0.f,0.f,0.f,0.f,0.f,0.f};
  v8f oacc[8];
  #pragma unroll
  for (int dn = 0; dn < 8; dn++) oacc[dn] = zero8;
  float run_m[8], run_l[8];
  #pragma unroll
  for (int r = 0; r < 8; r++) { run_m[r] = -__builtin_inff(); run_l[r] = 0.f; }

  const int kr = tid >> 3, c8 = tid & 7;
  const int vr = tid >> 1, vc = tid & 1;
  const float sscale = 0.08838834764831845f;   // 1/sqrt(128)

  // bb is always a literal -> constant LDS addresses
  auto issueKV = [&](int kb, int bb) {
    const __bf16* kp = Kh + (size_t)(kb * 32 + kr) * HEAD_DIM;
    async_ld_b128(lds_addr(&Ks[bb][kr * 136 + c8 * 8]), kp + c8 * 8);
    async_ld_b128(lds_addr(&Ks[bb][kr * 136 + 64 + c8 * 8]), kp + 64 + c8 * 8);
    const __bf16* vp = Vth + (size_t)vr * T_SEQ + kb * 32 + vc * 16;
    async_ld_b128(lds_addr(&Vts[bb][vr * 40 + vc * 16]), vp);
    async_ld_b128(lds_addr(&Vts[bb][vr * 40 + vc * 16 + 8]), vp + 8);
  };

  auto stepAttn = [&](int kb, int bb) {
    // ---- S = Q K^T (two 16x16 tiles: keys lm and 16+lm) ----
    v8f s0 = zero8, s1 = zero8;
    #pragma unroll
    for (int kc = 0; kc < 4; kc++) {
      bf16x16 t;
      const __bf16* k0p = &Ks[bb][lm * 136 + kc * 32 + half * 8];
      t.lo = *(const uint4*)k0p; t.hi = *(const uint4*)(k0p + 16);
      v16bf kf0 = __builtin_bit_cast(v16bf, t);
      const __bf16* k1p = &Ks[bb][(16 + lm) * 136 + kc * 32 + half * 8];
      t.lo = *(const uint4*)k1p; t.hi = *(const uint4*)(k1p + 16);
      v16bf kf1 = __builtin_bit_cast(v16bf, t);
      s0 = wmma_bf16(qf[kc], kf0, s0);
      s1 = wmma_bf16(qf[kc], kf1, s1);
    }
    #pragma unroll
    for (int r = 0; r < 8; r++) { s0[r] *= sscale; s1[r] *= sscale; }

    // ---- causal mask (C layout: row = qrow0 + 8*half + r, key = kb*32+n) ----
    if (kb * 32 + 31 > qrow0) {
      #pragma unroll
      for (int r = 0; r < 8; r++) {
        const int row  = qrow0 + 8 * half + r;
        const int key0 = kb * 32 + lm;
        if (key0 > row)      s0[r] = -__builtin_inff();
        if (key0 + 16 > row) s1[r] = -__builtin_inff();
      }
    }

    // ---- online softmax; write P (bf16) to wave-private LDS ----
    #pragma unroll
    for (int r = 0; r < 8; r++) {
      float mx = fmaxf(s0[r], s1[r]);
      #pragma unroll
      for (int off = 8; off >= 1; off >>= 1)
        mx = fmaxf(mx, __shfl_xor(mx, off, 32));   // reduce in 16-lane half
      const float nm = fmaxf(run_m[r], mx);
      const float sc = __expf(run_m[r] - nm);
      run_m[r] = nm;
      const float p0 = __expf(s0[r] - nm);
      const float p1 = __expf(s1[r] - nm);
      float rs = p0 + p1;
      #pragma unroll
      for (int off = 8; off >= 1; off >>= 1)
        rs += __shfl_xor(rs, off, 32);
      run_l[r] = run_l[r] * sc + rs;
      #pragma unroll
      for (int dn = 0; dn < 8; dn++) oacc[dn][r] *= sc;
      __bf16* pp = &Ps[wave][(r + 8 * half) * 40];
      pp[lm]      = f2bf(p0);
      pp[16 + lm] = f2bf(p1);
    }

    // ---- O += P @ V (A-layout P from LDS; B-layout Vt rows) ----
    const __bf16* pp = &Ps[wave][lm * 40 + half * 8];
    bf16x16 t; t.lo = *(const uint4*)pp; t.hi = *(const uint4*)(pp + 16);
    v16bf pf = __builtin_bit_cast(v16bf, t);
    #pragma unroll
    for (int dn = 0; dn < 8; dn++) {
      const __bf16* vpp = &Vts[bb][(dn * 16 + lm) * 40 + half * 8];
      bf16x16 tv; tv.lo = *(const uint4*)vpp; tv.hi = *(const uint4*)(vpp + 16);
      v16bf vf = __builtin_bit_cast(v16bf, tv);
      oacc[dn] = wmma_bf16(pf, vf, oacc[dn]);
    }
  };

  const int nkb = qb * 4 + 4;            // causal range; always even, >= 4
  issueKV(0, 0);
  issueKV(1, 1);
  for (int kb = 0; kb < nkb - 2; kb += 2) {
    wait_async<4>(); __syncthreads(); stepAttn(kb, 0);     __syncthreads();
    issueKV(kb + 2, 0);
    wait_async<4>(); __syncthreads(); stepAttn(kb + 1, 1); __syncthreads();
    issueKV(kb + 3, 1);
  }
  wait_async<4>(); __syncthreads(); stepAttn(nkb - 2, 0); __syncthreads();
  wait_async<0>(); __syncthreads(); stepAttn(nkb - 1, 1);

  // ---- normalize, store to attention output [B,T,C] (bf16) ----
  const int trow = qrow0 + 8 * half;
  #pragma unroll
  for (int r = 0; r < 8; r++) {
    const float inv = 1.f / run_l[r];
    const int t = trow + r;
    const size_t rowoff = ((size_t)b * T_SEQ + t) * D_MODEL + h * HEAD_DIM;
    #pragma unroll
    for (int dn = 0; dn < 8; dn++)
      AO[rowoff + dn * 16 + lm] = f2bf(oacc[dn][r] * inv);
  }
}

// ===========================================================================
extern "C" void kernel_launch(void* const* d_in, const int* in_sizes, int n_in,
                              void* d_out, int out_size, void* d_ws, size_t ws_size,
                              hipStream_t stream) {
  const float* x      = (const float*)d_in[0];   // [2,2048,2048]
  const float* W_qkv  = (const float*)d_in[1];   // [2048,6144]
  const float* b_qkv  = (const float*)d_in[2];   // [6144]
  const float* W_proj = (const float*)d_in[3];   // [2048,2048]
  const float* b_proj = (const float*)d_in[4];   // [2048]
  float* out = (float*)d_out;                    // [2,2048,2048]

  // workspace layout (bf16):
  const size_t SEG = (size_t)BT_ROWS * D_MODEL;         // 8,388,608 elems
  __bf16* Q     = (__bf16*)d_ws;                        // [B,H,T,hd]
  __bf16* K     = Q  + SEG;                             // [B,H,T,hd]
  __bf16* Vt    = K  + SEG;                             // [B,H,hd,T]
  __bf16* AO    = Vt + SEG;                             // [4096,2048]
  __bf16* Xb    = AO + SEG;                             // x as bf16
  __bf16* Wqt   = Xb + SEG;                             // W_qkv^T  [6144][2048]
  __bf16* Wpt   = Wqt + 3 * SEG;                        // W_proj^T [2048][2048]
  (void)in_sizes; (void)n_in; (void)out_size; (void)ws_size;

  // pre-pass: conversions / transposes (bandwidth-trivial)
  cvt_bf16_kernel<<<dim3(SEG / (256 * 8)), 256, 0, stream>>>(x, Xb);
  cvt_transpose_kernel<<<dim3(6144 / 32, D_MODEL / 32), 256, 0, stream>>>(
      W_qkv, Wqt, D_MODEL, 6144);
  cvt_transpose_kernel<<<dim3(2048 / 32, D_MODEL / 32), 256, 0, stream>>>(
      W_proj, Wpt, D_MODEL, 2048);

  qkv_gemm_kernel<<<dim3(6144 / 256, BT_ROWS / 64), 256, 0, stream>>>(
      Xb, Wqt, b_qkv, Q, K, Vt);
  attn_kernel<<<dim3(T_SEQ / 128, N_HEADS, 2), 256, 0, stream>>>(Q, K, Vt, AO);
  proj_gemm_kernel<<<dim3(D_MODEL / 256, BT_ROWS / 64), 256, 0, stream>>>(
      AO, Wpt, b_proj, out);
}
